// BiUTE_26594437496904
// MI455X (gfx1250) — compile-verified, matrix operand order checked by reading, not run on hardware
//
#include <hip/hip_runtime.h>
#include <cstdint>

typedef __bf16 bf16_t;
typedef bf16_t bf16x8  __attribute__((ext_vector_type(8)));
typedef bf16_t bf16x16 __attribute__((ext_vector_type(16)));
typedef float  floatx8 __attribute__((ext_vector_type(8)));

#define BM 128
#define BN 128
#define BKK 32
#define LDS_STRIDE 40   // 32 + 8 pad: 80B row stride, 16B aligned, conflict-free frag reads
#define BUFELEMS (BM * LDS_STRIDE)   // 5120 bf16 per buffer
#define BUFBYTES (BUFELEMS * 2)      // 10240 bytes

// ---------------------------------------------------------------------------
// problem sizes
#define BSZ 16
#define TLEN 128
#define NP 12
#define DIM 1024
#define D2 2048
#define MROWS (BSZ * TLEN)   // 2048

// workspace layout (bytes)
#define OFF_WQSUM  ((size_t)0)
#define OFF_NF32   ((size_t)4096)
#define OFF_NBF    (OFF_NF32 + (size_t)MROWS * D2 * 4)       // 16 MB
#define OFF_W1     (OFF_NBF  + (size_t)MROWS * D2 * 2)       // 8 MB
#define OFF_W2     (OFF_W1   + (size_t)4096 * D2 * 2)        // 16 MB
#define OFF_PROJ   (OFF_W2   + (size_t)4096 * D2 * 2)        // 16 MB
#define OFF_GBT    (OFF_PROJ + (size_t)MROWS * 4096 * 2)     // 16 MB
#define OFF_S      (OFF_GBT  + (size_t)4096 * MROWS * 2)     // 16 MB
#define OFF_OB     (OFF_S    + (size_t)32 * 128 * 128 * 2)   // 1 MB
#define OFF_OA     (OFF_OB   + (size_t)MROWS * D2 * 4)       // 16 MB

// ---------------------------------------------------------------------------
// CDNA5 async global->LDS copy (16B per lane), tracked by ASYNCcnt.
__device__ __forceinline__ void async_cp16(uint32_t lds_addr, const void* gaddr) {
  asm volatile("global_load_async_to_lds_b128 %0, %1, off"
               :: "v"(lds_addr), "v"(gaddr)
               : "memory");
}
__device__ __forceinline__ void wait_async4() {
  asm volatile("s_wait_asynccnt 0x4" ::: "memory");
}
__device__ __forceinline__ void wait_async0() {
  asm volatile("s_wait_asynccnt 0x0" ::: "memory");
}
// low 32 bits of a generic pointer to LDS == LDS byte offset (shared aperture)
__device__ __forceinline__ uint32_t lds_addr_of(const void* p) {
  return (uint32_t)(uint64_t)p;
}

// ---------------------------------------------------------------------------
// fragment loader: 16 bf16 per lane as two contiguous 16B chunks.
// ISA 16-bit A/B layout: lane half h -> chunk0 at K=8h, chunk1 at K=16+8h.
__device__ __forceinline__ bf16x16 load_frag(const bf16_t* row_base, int hsel) {
  bf16x8 a = *(const bf16x8*)(row_base + 8 * hsel);
  bf16x8 b = *(const bf16x8*)(row_base + 16 + 8 * hsel);
  return __builtin_shufflevector(a, b, 0, 1, 2, 3, 4, 5, 6, 7,
                                       8, 9, 10, 11, 12, 13, 14, 15);
}

// Shared WMMA GEMM core: C += A[M,K] * B^T where Bm is [N,K] row-major.
// Block tile 128x128, 8 waves in 4(M) x 2(N), wave tile 32x64 (2x4 frags),
// BK=32, double-buffered LDS fed by async global->LDS DMA.
__device__ __forceinline__ void gemm_core(const bf16_t* __restrict__ A, int lda,
                                          const bf16_t* __restrict__ Bm, int ldb,
                                          int K, floatx8 acc[2][4],
                                          bf16_t* As, bf16_t* Bs) {
  const int tid  = threadIdx.x;
  const int lane = tid & 31;
  const int wave = tid >> 5;
  const int wm   = wave & 3;     // 4 wave-rows of 32
  const int wn   = wave >> 2;    // 2 wave-cols of 64
  const int lrow = lane & 15;
  const int hsel = lane >> 4;

  const int ldRow = tid >> 1;          // 0..127
  const int ldCol = (tid & 1) * 16;    // 0 or 16

  const uint32_t ldsA = lds_addr_of(As) + (uint32_t)((ldRow * LDS_STRIDE + ldCol) * 2);
  const uint32_t ldsB = lds_addr_of(Bs) + (uint32_t)((ldRow * LDS_STRIDE + ldCol) * 2);
  const bf16_t* gA = A + (size_t)ldRow * lda + ldCol;
  const bf16_t* gB = Bm + (size_t)ldRow * ldb + ldCol;

  const int nsteps = K / BKK;

  // prologue: stage 0 -> buffer 0 (4 async B128 per thread)
  async_cp16(ldsA, gA);      async_cp16(ldsA + 16, gA + 8);
  async_cp16(ldsB, gB);      async_cp16(ldsB + 16, gB + 8);

  for (int step = 0; step < nsteps; ++step) {
    const int buf = step & 1;
    if (step + 1 < nsteps) {
      const int k1 = (step + 1) * BKK;
      const uint32_t boff = (uint32_t)((step + 1) & 1) * BUFBYTES;
      async_cp16(ldsA + boff, gA + k1);      async_cp16(ldsA + boff + 16, gA + k1 + 8);
      async_cp16(ldsB + boff, gB + k1);      async_cp16(ldsB + boff + 16, gB + k1 + 8);
      wait_async4();   // async loads retire in order: current stage is resident
    } else {
      wait_async0();
    }
    __syncthreads();

    const bf16_t* Ab = As + buf * BUFELEMS;
    const bf16_t* Bb = Bs + buf * BUFELEMS;
    bf16x16 afrag[2], bfrag[4];
#pragma unroll
    for (int mt = 0; mt < 2; ++mt) {
      int r = wm * 32 + mt * 16 + lrow;
      afrag[mt] = load_frag(&Ab[r * LDS_STRIDE], hsel);
    }
#pragma unroll
    for (int nt = 0; nt < 4; ++nt) {
      int c = wn * 64 + nt * 16 + lrow;
      bfrag[nt] = load_frag(&Bb[c * LDS_STRIDE], hsel);
    }
#pragma unroll
    for (int mt = 0; mt < 2; ++mt)
#pragma unroll
      for (int nt = 0; nt < 4; ++nt)
        acc[mt][nt] = __builtin_amdgcn_wmma_f32_16x16x32_bf16(
            false, afrag[mt], false, bfrag[nt], (short)0, acc[mt][nt],
            false, false);
    __syncthreads();   // protects buffer reuse by stage step+2 prefetch
  }
}

// ---------------------------------------------------------------------------
// 0) wqsum[d] = sum_n Wq[n,d]
__global__ void wqsum_kernel(const float* __restrict__ Wq,
                             float* __restrict__ wqsum) {
  int d = blockIdx.x * blockDim.x + threadIdx.x;
  if (d < DIM) {
    float s = 0.f;
#pragma unroll
    for (int n = 0; n < NP; ++n) s += Wq[n * DIM + d];
    wqsum[d] = s;
  }
}

// 1) n = concat(g, f) per (b,t); emit f32 (residual) + bf16 (GEMM operand)
__global__ void __launch_bounds__(256)
build_n_kernel(const float* __restrict__ feats, const float* __restrict__ wqsum,
               float* __restrict__ n_f32, bf16_t* __restrict__ n_bf16) {
  __shared__ float feat[NP * DIM];     // 48 KB
  __shared__ float red[8 * NP];
  __shared__ float wsh[NP];
  const int row = blockIdx.x;          // b*T + t
  const int tid = threadIdx.x;

  // async DMA the 12x1024 feature slab straight into LDS
  const float4* src = (const float4*)(feats + (size_t)row * NP * DIM);
  const uint32_t ldsfeat = lds_addr_of(feat);
  for (int i = tid; i < NP * (DIM / 4); i += 256)
    async_cp16(ldsfeat + (uint32_t)i * 16, src + i);
  wait_async0();
  __syncthreads();

  const float4 wq = ((const float4*)wqsum)[tid];   // d = tid*4 .. tid*4+3
  float part[NP];
#pragma unroll
  for (int n = 0; n < NP; ++n) {
    float4 fv = ((const float4*)feat)[n * (DIM / 4) + tid];
    part[n] = fv.x * wq.x + fv.y * wq.y + fv.z * wq.z + fv.w * wq.w;
  }
#pragma unroll
  for (int n = 0; n < NP; ++n) {
    float v = part[n];
    for (int off = 16; off > 0; off >>= 1) v += __shfl_xor(v, off, 32);
    part[n] = v;
  }
  const int lane = tid & 31, wave = tid >> 5;
  if (lane == 0)
#pragma unroll
    for (int n = 0; n < NP; ++n) red[wave * NP + n] = part[n];
  __syncthreads();
  if (tid < NP) {
    float s = 0.f;
#pragma unroll
    for (int w = 0; w < 8; ++w) s += red[w * NP + tid];
    wsh[tid] = s;
  }
  __syncthreads();

  float4 g = {0.f, 0.f, 0.f, 0.f};
  float4 fm = {-3.4e38f, -3.4e38f, -3.4e38f, -3.4e38f};
#pragma unroll
  for (int n = 0; n < NP; ++n) {
    float4 fv = ((const float4*)feat)[n * (DIM / 4) + tid];
    float wn = wsh[n];
    g.x += wn * fv.x; g.y += wn * fv.y; g.z += wn * fv.z; g.w += wn * fv.w;
    fm.x = fmaxf(fm.x, fv.x); fm.y = fmaxf(fm.y, fv.y);
    fm.z = fmaxf(fm.z, fv.z); fm.w = fmaxf(fm.w, fv.w);
  }
  float* nrow = n_f32 + (size_t)row * D2;
  ((float4*)nrow)[tid] = g;
  ((float4*)(nrow + DIM))[tid] = fm;
  bf16_t* brow = n_bf16 + (size_t)row * D2;
  int d = tid * 4;
  brow[d + 0] = (bf16_t)g.x;  brow[d + 1] = (bf16_t)g.y;
  brow[d + 2] = (bf16_t)g.z;  brow[d + 3] = (bf16_t)g.w;
  brow[DIM + d + 0] = (bf16_t)fm.x; brow[DIM + d + 1] = (bf16_t)fm.y;
  brow[DIM + d + 2] = (bf16_t)fm.z; brow[DIM + d + 3] = (bf16_t)fm.w;
}

// 2) f32 -> bf16 cast (weight concatenation)
__global__ void cvt_kernel(const float* __restrict__ src,
                           bf16_t* __restrict__ dst, int count) {
  for (int i = blockIdx.x * blockDim.x + threadIdx.x; i < count;
       i += gridDim.x * blockDim.x)
    dst[i] = (bf16_t)src[i];
}

// 3) projection GEMM: C[M,N] = n * Wcat^T   (bf16 out; optionally transposed)
template <bool TRSTORE>
__global__ void __launch_bounds__(256)
proj_gemm_kernel(const bf16_t* __restrict__ A, const bf16_t* __restrict__ W,
                 bf16_t* __restrict__ C, int M, int N, int K) {
  __shared__ bf16_t As[2 * BUFELEMS];
  __shared__ bf16_t Bs[2 * BUFELEMS];
  const int m0 = blockIdx.y * BM, n0 = blockIdx.x * BN;
  floatx8 acc[2][4] = {};
  gemm_core(A + (size_t)m0 * K, K, W + (size_t)n0 * K, K, K, acc, As, Bs);

  const int lane = threadIdx.x & 31, wave = threadIdx.x >> 5;
  const int wm = wave & 3, wn = wave >> 2;
  const int lrow = lane & 15, hsel = lane >> 4;
#pragma unroll
  for (int mt = 0; mt < 2; ++mt)
#pragma unroll
    for (int nt = 0; nt < 4; ++nt) {
      int nl = wn * 64 + nt * 16 + lrow;
      if (TRSTORE) {
        // C^T: per lane the 8 accum rows are contiguous -> one 16B store
        bf16x8 v;
#pragma unroll
        for (int r = 0; r < 8; ++r) v[r] = (bf16_t)acc[mt][nt][r];
        *(bf16x8*)(C + (size_t)(n0 + nl) * M +
                   (m0 + wm * 32 + mt * 16 + 8 * hsel)) = v;
      } else {
#pragma unroll
        for (int r = 0; r < 8; ++r) {
          int ml = wm * 32 + mt * 16 + r + 8 * hsel;
          C[(size_t)(m0 + ml) * N + (n0 + nl)] = (bf16_t)acc[mt][nt][r];
        }
      }
    }
}

// 4) scores: S = (t . p^T) * scale, strict-triangular mask, bf16 out
__global__ void __launch_bounds__(256)
score_kernel(const bf16_t* __restrict__ proj, bf16_t* __restrict__ S,
             float scale) {
  __shared__ bf16_t As[2 * BUFELEMS];
  __shared__ bf16_t Bs[2 * BUFELEMS];
  const int z = blockIdx.x;        // b*2 + branch
  const int b = z >> 1, br = z & 1;
  const bf16_t* A  = proj + (size_t)(b * TLEN) * 4096 + (br ? 2048 : 0);
  const bf16_t* Bm = proj + (size_t)(b * TLEN) * 4096 + (br ? 3072 : 1024);
  floatx8 acc[2][4] = {};
  gemm_core(A, 4096, Bm, 4096, DIM, acc, As, Bs);

  const int lane = threadIdx.x & 31, wave = threadIdx.x >> 5;
  const int wm = wave & 3, wn = wave >> 2;
  const int lrow = lane & 15, hsel = lane >> 4;
  bf16_t* Sp = S + (size_t)z * TLEN * TLEN;
#pragma unroll
  for (int mt = 0; mt < 2; ++mt)
#pragma unroll
    for (int nt = 0; nt < 4; ++nt)
#pragma unroll
      for (int r = 0; r < 8; ++r) {
        int i = wm * 32 + mt * 16 + r + 8 * hsel;
        int j = wn * 64 + nt * 16 + lrow;
        bool keep = br ? (j > i) : (j < i);
        float v = keep ? acc[mt][nt][r] * scale : 0.f;
        Sp[i * TLEN + j] = (bf16_t)v;
      }
}

// 5) out = S * g   (B operand is gbT: [feature, m] row-major, K-contiguous)
__global__ void __launch_bounds__(256)
out_gemm_kernel(const bf16_t* __restrict__ S, const bf16_t* __restrict__ gbT,
                float* __restrict__ ob, float* __restrict__ oa) {
  __shared__ bf16_t As[2 * BUFELEMS];
  __shared__ bf16_t Bs[2 * BUFELEMS];
  const int z = blockIdx.z;        // b*2 + branch
  const int b = z >> 1, br = z & 1;
  const int f0 = blockIdx.x * BN;
  const bf16_t* A  = S + (size_t)z * TLEN * TLEN;                        // lda 128
  const bf16_t* Bm = gbT + ((size_t)(br ? D2 : 0) + f0) * MROWS + b * TLEN;
  floatx8 acc[2][4] = {};
  gemm_core(A, TLEN, Bm, MROWS, TLEN, acc, As, Bs);

  const int lane = threadIdx.x & 31, wave = threadIdx.x >> 5;
  const int wm = wave & 3, wn = wave >> 2;
  const int lrow = lane & 15, hsel = lane >> 4;
  float* O = (br ? oa : ob) + (size_t)(b * TLEN) * D2 + f0;
#pragma unroll
  for (int mt = 0; mt < 2; ++mt)
#pragma unroll
    for (int nt = 0; nt < 4; ++nt)
#pragma unroll
      for (int r = 0; r < 8; ++r) {
        int ml = wm * 32 + mt * 16 + r + 8 * hsel;
        int nl = wn * 64 + nt * 16 + lrow;
        O[(size_t)ml * D2 + nl] = acc[mt][nt][r];
      }
}

// 6) residual add
__global__ void final_add_kernel(const float* __restrict__ nf,
                                 const float* __restrict__ ob,
                                 const float* __restrict__ oa,
                                 float* __restrict__ out, int count) {
  for (int i = blockIdx.x * blockDim.x + threadIdx.x; i < count;
       i += gridDim.x * blockDim.x)
    out[i] = nf[i] + ob[i] + oa[i];
}

// ---------------------------------------------------------------------------
extern "C" void kernel_launch(void* const* d_in, const int* in_sizes, int n_in,
                              void* d_out, int out_size, void* d_ws,
                              size_t ws_size, hipStream_t stream) {
  (void)in_sizes; (void)n_in; (void)out_size; (void)ws_size;
  const float* feats = (const float*)d_in[0];
  const float* Wq    = (const float*)d_in[1];
  const float* Wtb   = (const float*)d_in[2];
  const float* Wpb   = (const float*)d_in[3];
  const float* Wgb   = (const float*)d_in[4];
  const float* Wta   = (const float*)d_in[5];
  const float* Wpa   = (const float*)d_in[6];
  const float* Wga   = (const float*)d_in[7];

  char* ws = (char*)d_ws;
  float*  wqsum  = (float*)(ws + OFF_WQSUM);
  float*  n_f32  = (float*)(ws + OFF_NF32);
  bf16_t* n_bf16 = (bf16_t*)(ws + OFF_NBF);
  bf16_t* Wcat1  = (bf16_t*)(ws + OFF_W1);    // [Wtb;Wpb;Wta;Wpa]  4096 x 2048
  bf16_t* Wcat2  = (bf16_t*)(ws + OFF_W2);    // [Wgb;Wga]          4096 x 2048
  bf16_t* proj   = (bf16_t*)(ws + OFF_PROJ);  // 2048 x 4096 (tb|pb|ta|pa)
  bf16_t* gbT    = (bf16_t*)(ws + OFF_GBT);   // 4096 x 2048 (gb^T ; ga^T)
  bf16_t* Sbuf   = (bf16_t*)(ws + OFF_S);     // 32 x 128 x 128
  float*  ob     = (float*)(ws + OFF_OB);
  float*  oa     = (float*)(ws + OFF_OA);

  wqsum_kernel<<<4, 256, 0, stream>>>(Wq, wqsum);
  build_n_kernel<<<MROWS, 256, 0, stream>>>(feats, wqsum, n_f32, n_bf16);

  const int wsmall = DIM * D2;   // 2,097,152
  const int wbig   = D2 * D2;    // 4,194,304
  cvt_kernel<<<2048, 256, 0, stream>>>(Wtb, Wcat1 + 0 * (size_t)wsmall, wsmall);
  cvt_kernel<<<2048, 256, 0, stream>>>(Wpb, Wcat1 + 1 * (size_t)wsmall, wsmall);
  cvt_kernel<<<2048, 256, 0, stream>>>(Wta, Wcat1 + 2 * (size_t)wsmall, wsmall);
  cvt_kernel<<<2048, 256, 0, stream>>>(Wpa, Wcat1 + 3 * (size_t)wsmall, wsmall);
  cvt_kernel<<<2048, 256, 0, stream>>>(Wgb, Wcat2 + 0 * (size_t)wbig, wbig);
  cvt_kernel<<<2048, 256, 0, stream>>>(Wga, Wcat2 + 1 * (size_t)wbig, wbig);

  // projections: M=2048, N=4096, K=2048 each
  proj_gemm_kernel<false><<<dim3(32, 16), 256, 0, stream>>>(
      n_bf16, Wcat1, proj, MROWS, 4096, D2);
  proj_gemm_kernel<true><<<dim3(32, 16), 256, 0, stream>>>(
      n_bf16, Wcat2, gbT, MROWS, 4096, D2);

  const float scale = 1.0f / sqrtf((float)D2);
  score_kernel<<<32, 256, 0, stream>>>(proj, Sbuf, scale);
  out_gemm_kernel<<<dim3(16, 1, 32), 256, 0, stream>>>(Sbuf, gbT, ob, oa);

  final_add_kernel<<<4096, 256, 0, stream>>>(n_f32, ob, oa, (float*)d_out,
                                             MROWS * D2);
}
